// DeepNet_6408091205736
// MI455X (gfx1250) — matrix operand order repaired and checked by reading, not compiled, not run on hardware
//
#include <hip/hip_runtime.h>

// ---------------- Problem constants (match reference) ----------------
#define NNODES 100000
#define NEDGES 1000000
#define DIM    128
#define NREL   8
#define KTOT   ((NREL + 1) * DIM)   // 1152: 8 relation slots + root (read from x)
#define KSTEPS (KTOT / 32)          // 36 WMMA K-steps
#define NTILES (DIM / 16)           // 8 WMMA N-tiles

typedef __bf16 bf16_t;
typedef bf16_t          v16bf __attribute__((ext_vector_type(16)));
typedef bf16_t          v8bf  __attribute__((ext_vector_type(8)));
typedef bf16_t          v4bf  __attribute__((ext_vector_type(4)));
typedef float           v8f   __attribute__((ext_vector_type(8)));
typedef unsigned short  v16u  __attribute__((ext_vector_type(16)));

// f32 -> bf16 bits, round-to-nearest-even, pure integer ops
__device__ __forceinline__ unsigned short f2bf_bits(float f) {
    union { float f; unsigned u; } a; a.f = f;
    unsigned u = a.u;
    u += 0x7FFFu + ((u >> 16) & 1u);
    return (unsigned short)(u >> 16);
}
// bf16 bits (in low 16 of u) -> f32
__device__ __forceinline__ float bf2f(unsigned u) {
    union { unsigned u; float f; } a; a.u = u << 16;
    return a.f;
}

// packed bf16 global atomic add: CDNA5 GLOBAL_ATOMIC_PK_ADD_BF16
__device__ __forceinline__ void pk_add_bf16(unsigned* addr, float lo, float hi) {
    unsigned packed = (unsigned)f2bf_bits(lo) | ((unsigned)f2bf_bits(hi) << 16);
#if __has_builtin(__builtin_amdgcn_global_atomic_fadd_v2bf16)
    typedef short v2s __attribute__((ext_vector_type(2)));
    typedef v2s __attribute__((address_space(1))) v2s_g;
    union { unsigned u; v2s s; } pv; pv.u = packed;
    (void)__builtin_amdgcn_global_atomic_fadd_v2bf16(
        (v2s_g*)(unsigned long long)(uintptr_t)addr, pv.s);
#else
    asm volatile("global_atomic_pk_add_bf16 %0, %1, off"
                 :: "v"((unsigned long long)(uintptr_t)addr), "v"(packed)
                 : "memory");
#endif
}

// ---------------- Degree / normalization ----------------
__global__ void k_degree(const int* __restrict__ dst, const int* __restrict__ et,
                         float* __restrict__ deg) {
    int e = blockIdx.x * blockDim.x + threadIdx.x;
    if (e >= NEDGES) return;
    atomicAdd(&deg[(size_t)et[e] * NNODES + dst[e]], 1.0f);
}

__global__ void k_inv(float* __restrict__ deg) {
    int i = blockIdx.x * blockDim.x + threadIdx.x;
    if (i >= NREL * NNODES) return;
    float d = deg[i];
    deg[i] = (d > 0.0f) ? (1.0f / d) : 0.0f;
}

// ---------------- Pack W (+root) into bf16 WMMA B-fragments ----------------
// Fragment f = kt*8+nt is 32(K)x16(N). Lane l<16: col n, Klocal 0..7,16..23;
// lane l+16: Klocal 8..15,24..31. b64-granular lane-major storage:
//   idx = ((f*4 + h/4)*32 + lane)*4 + h%4
// -> ds_load_b64 lane stride 8B = 2 dwords -> all 64 LDS banks hit once.
__global__ void k_pack(const float* __restrict__ W, const float* __restrict__ root,
                       unsigned short* __restrict__ Wp) {
    int i = blockIdx.x * blockDim.x + threadIdx.x;   // over KTOT*DIM = 147456
    if (i >= KTOT * DIM) return;
    int k = i / DIM, n = i % DIM;
    float v = (k < NREL * DIM) ? W[(size_t)k * DIM + n]
                               : root[(size_t)(k - NREL * DIM) * DIM + n];
    int kt = k >> 5, kl = k & 31, nt = n >> 4, nl = n & 15;
    int lane = ((kl >> 3) & 1) ? (nl + 16) : nl;
    int h = (kl & 7) | ((kl >> 4) << 3);
    int f = kt * NTILES + nt;
    Wp[((size_t)(f * 4 + (h >> 2)) * 32 + lane) * 4 + (h & 3)] = f2bf_bits(v);
}

// ---------------- Weighted scatter: A[v][etype][:] += inv_deg * x[src][:] ----------------
// A is bf16 now: 32 threads/edge, each owns 4 features -> 2 pk_add_bf16 atomics.
template <bool LAYER1>
__global__ void k_scatter(const void* __restrict__ xin, const int* __restrict__ src,
                          const int* __restrict__ dst, const int* __restrict__ et,
                          const float* __restrict__ inv, unsigned* __restrict__ Ab) {
    long long idx = (long long)blockIdx.x * blockDim.x + threadIdx.x;
    if (idx >= (long long)NEDGES * 32) return;
    int e = (int)(idx >> 5);
    int t = (int)(idx & 31);
    int r = et[e], v = dst[e], s = src[e];
    float w = inv[(size_t)r * NNODES + v];
    float a0, a1, a2, a3;
    if constexpr (LAYER1) {
        const float4 xs = *(const float4*)((const float*)xin + (size_t)s * DIM + t * 4);
        a0 = xs.x * w; a1 = xs.y * w; a2 = xs.z * w; a3 = xs.w * w;
    } else {
        const uint2 hx = *(const uint2*)((const unsigned short*)xin + (size_t)s * DIM + t * 4);
        a0 = bf2f(hx.x & 0xFFFFu) * w; a1 = bf2f(hx.x >> 16) * w;
        a2 = bf2f(hx.y & 0xFFFFu) * w; a3 = bf2f(hx.y >> 16) * w;
    }
    unsigned* ap = Ab + (((size_t)v * NREL + r) * DIM + t * 4) / 2;
    pk_add_bf16(ap + 0, a0, a1);
    pk_add_bf16(ap + 1, a2, a3);
}

// ---------------- Fused GEMM + bias + ReLU ----------------
// out[v,:] = relu( [A[v] | x[v]] @ Wcat + b ), K = 1152, A already bf16.
// Block = 256 threads = 8 waves; wave w computes rows [blk*128+w*16, +16) x 128 cols.
// Wcat (288 KB bf16) staged once per block in LDS (320 KB/WGP); conflict-free b64 reads.
template <bool LAYER1>   // LAYER1: x is f32, out is bf16; else x is bf16, out is f32
__global__ __launch_bounds__(256)
void k_gemm(const unsigned short* __restrict__ Ab, const void* __restrict__ xin,
            const unsigned short* __restrict__ Wp,
            const float* __restrict__ bias, void* __restrict__ outp) {
    __shared__ unsigned short sW[KTOT * DIM];    // 294912 B of 320 KB WGP LDS
    const int tid = threadIdx.x;
    {
        const uint4* gw = (const uint4*)Wp;
        uint4* lw = (uint4*)sW;
        #pragma unroll 4
        for (int i = tid; i < (KTOT * DIM) / 8; i += 256) lw[i] = gw[i];
    }
    __syncthreads();

    const int wave = tid >> 5;
    const int lane = tid & 31;
    const int grp  = lane >> 4;     // 0: K 0..7 / 16..23   1: K 8..15 / 24..31
    const int lr   = lane & 15;
    const int rowBase = blockIdx.x * 128 + wave * 16;
    int arow = rowBase + lr;
    if (arow >= NNODES) arow = NNODES - 1;          // clamp loads, keep EXEC full
    const unsigned short* ap = Ab + (size_t)arow * (NREL * DIM) + grp * 8;
    const v4bf* lw4 = (const v4bf*)sW;

    v8f acc[NTILES] = {};

    auto mm = [&](const v16bf& avv, int kt) {
        #pragma unroll
        for (int nt = 0; nt < NTILES; ++nt) {
            const int f = kt * NTILES + nt;
            union { v4bf c[4]; v16bf v; } bv;
            #pragma unroll
            for (int c = 0; c < 4; ++c)
                bv.c[c] = lw4[(size_t)(f * 4 + c) * 32 + lane];
            acc[nt] = __builtin_amdgcn_wmma_f32_16x16x32_bf16(
                false, avv, false, bv.v, (short)0, acc[nt], false, false);
        }
    };
    auto stepBF = [&](const unsigned short* bp, int kt) {
        union { v8bf h[2]; v16bf v; } av;
        av.h[0] = *(const v8bf*)(bp);       // 16B: K chunk lo
        av.h[1] = *(const v8bf*)(bp + 16);  // 16B: K chunk hi
        mm(av.v, kt);
    };

    // relation K-blocks stream straight from bf16 A (no conversion in hot loop)
    for (int kt = 0; kt < KSTEPS - 4; ++kt) {
        if (kt + 8 < KSTEPS - 4)
            __builtin_prefetch(ap + (kt + 8) * 32, 0, 1);   // global_prefetch_b8
        stepBF(ap + kt * 32, kt);
    }
    // root K-block reads x directly
    if constexpr (LAYER1) {
        const float* xp = (const float*)xin + (size_t)arow * DIM + grp * 8;
        #pragma unroll
        for (int kt = KSTEPS - 4; kt < KSTEPS; ++kt) {
            const float* bp = xp + (kt - (KSTEPS - 4)) * 32;
            const float4 f0 = *(const float4*)(bp);
            const float4 f1 = *(const float4*)(bp + 4);
            const float4 f2 = *(const float4*)(bp + 16);
            const float4 f3 = *(const float4*)(bp + 20);
            union { v16u u; v16bf b; } av;
            av.u[0]  = f2bf_bits(f0.x); av.u[1]  = f2bf_bits(f0.y);
            av.u[2]  = f2bf_bits(f0.z); av.u[3]  = f2bf_bits(f0.w);
            av.u[4]  = f2bf_bits(f1.x); av.u[5]  = f2bf_bits(f1.y);
            av.u[6]  = f2bf_bits(f1.z); av.u[7]  = f2bf_bits(f1.w);
            av.u[8]  = f2bf_bits(f2.x); av.u[9]  = f2bf_bits(f2.y);
            av.u[10] = f2bf_bits(f2.z); av.u[11] = f2bf_bits(f2.w);
            av.u[12] = f2bf_bits(f3.x); av.u[13] = f2bf_bits(f3.y);
            av.u[14] = f2bf_bits(f3.z); av.u[15] = f2bf_bits(f3.w);
            mm(av.b, kt);
        }
    } else {
        const unsigned short* xp = (const unsigned short*)xin + (size_t)arow * DIM + grp * 8;
        #pragma unroll
        for (int kt = KSTEPS - 4; kt < KSTEPS; ++kt)
            stepBF(xp + (kt - (KSTEPS - 4)) * 32, kt);
    }

    // C/D layout: VGPR g -> row M = g + grp*8; col N = lane&15 within tile
    const int row0 = rowBase + grp * 8;
    #pragma unroll
    for (int nt = 0; nt < NTILES; ++nt) {
        const int n = nt * 16 + lr;
        const float bb = bias[n];
        #pragma unroll
        for (int g = 0; g < 8; ++g) {
            int row = row0 + g;
            if (row < NNODES) {
                float val = fmaxf(acc[nt][g] + bb, 0.0f);
                if constexpr (LAYER1)
                    ((unsigned short*)outp)[(size_t)row * DIM + n] = f2bf_bits(val);
                else
                    ((float*)outp)[(size_t)row * DIM + n] = val;
            }
        }
    }
}

// ---------------- Launcher ----------------
extern "C" void kernel_launch(void* const* d_in, const int* in_sizes, int n_in,
                              void* d_out, int out_size, void* d_ws, size_t ws_size,
                              hipStream_t stream) {
    const float* x  = (const float*)d_in[0];
    const int*   ei = (const int*)d_in[1];
    const int*   et = (const int*)d_in[2];
    const float* W1 = (const float*)d_in[3];
    const float* r1 = (const float*)d_in[4];
    const float* b1 = (const float*)d_in[5];
    const float* W2 = (const float*)d_in[6];
    const float* r2 = (const float*)d_in[7];
    const float* b2 = (const float*)d_in[8];
    const int* src = ei;
    const int* dst = ei + NEDGES;

    // workspace carve-up (all offsets 256B aligned)
    const size_t bytesA   = (size_t)NNODES * NREL * DIM * sizeof(unsigned short); // 204.8 MB
    const size_t bytesDeg = (size_t)NREL * NNODES * sizeof(float);                // 3.2 MB
    const size_t bytesWp  = (size_t)KTOT * DIM * sizeof(unsigned short);          // 288 KB
    char* ws = (char*)d_ws;
    unsigned short* Ab  = (unsigned short*)ws;
    float*          deg = (float*)(ws + bytesA);
    unsigned short* Wp  = (unsigned short*)(ws + bytesA + bytesDeg);
    unsigned short* h1  = (unsigned short*)(ws + bytesA + bytesDeg + bytesWp);    // 25.6 MB

    const int B = 256;
    const int gDeg  = (NEDGES + B - 1) / B;
    const int gInv  = (NREL * NNODES + B - 1) / B;
    const int gPack = (KTOT * DIM + B - 1) / B;
    const int gScat = (int)(((long long)NEDGES * 32 + B - 1) / B);
    const int gGemm = (NNODES + 127) / 128;

    // degrees are layer-invariant: compute once
    hipMemsetAsync(deg, 0, bytesDeg, stream);
    k_degree<<<gDeg, B, 0, stream>>>(dst, et, deg);
    k_inv<<<gInv, B, 0, stream>>>(deg);

    // ---- layer 1: x (f32) -> h1 (bf16) ----
    hipMemsetAsync(Ab, 0, bytesA, stream);
    k_pack<<<gPack, B, 0, stream>>>(W1, r1, Wp);
    k_scatter<true><<<gScat, B, 0, stream>>>(x, src, dst, et, deg, (unsigned*)Ab);
    k_gemm<true><<<gGemm, B, 0, stream>>>(Ab, x, Wp, b1, h1);

    // ---- layer 2: h1 (bf16) -> d_out (f32) ----
    hipMemsetAsync(Ab, 0, bytesA, stream);
    k_pack<<<gPack, B, 0, stream>>>(W2, r2, Wp);
    k_scatter<false><<<gScat, B, 0, stream>>>(h1, src, dst, et, deg, (unsigned*)Ab);
    k_gemm<false><<<gGemm, B, 0, stream>>>(Ab, h1, Wp, b2, d_out);
}
